// SC_attn_79310866088423
// MI455X (gfx1250) — matrix operand order repaired and checked by reading, not compiled, not run on hardware
//
#include <hip/hip_runtime.h>
#include <hip/hip_bf16.h>
#include <math.h>

#define N_ROWS 8192
#define IN_CH  512
#define OUT_CH 512

typedef __attribute__((ext_vector_type(16))) __bf16 v16bf;
typedef __attribute__((ext_vector_type(8)))  __bf16 v8bf;
typedef __attribute__((ext_vector_type(8)))  float  v8f;

// ---------- helpers ----------

__device__ __forceinline__ __bf16 f2bf(float f) {
    union { float f; unsigned u; } a; a.f = f;
    unsigned r = a.u + 0x7FFFu + ((a.u >> 16) & 1u);   // round-to-nearest-even
    unsigned short h = (unsigned short)(r >> 16);
    __bf16 o; __builtin_memcpy(&o, &h, sizeof(o));
    return o;
}

__device__ __forceinline__ v16bf cat8(v8bf lo, v8bf hi) {
    return __builtin_shufflevector(lo, hi, 0,1,2,3,4,5,6,7,8,9,10,11,12,13,14,15);
}

// 16-bit A fragment (16x32, MxK), wave32 layout (ISA 7.12.2):
//   lane = M + 16*hk ; elems 0..7 = K[hk*8..+7] ; elems 8..15 = K[16+hk*8..+7]
__device__ __forceinline__ v16bf load_a_frag(const __bf16* row, int hk) {
    v8bf lo = *(const v8bf*)(row + hk * 8);
    v8bf hi = *(const v8bf*)(row + 16 + hk * 8);
    return cat8(lo, hi);
}

// 16-bit B fragment (32x16, KxN): lane supplies column n = lane&15, hk = lane>>4,
// elems 0..7 = K[hk*8..+7] ; elems 8..15 = K[16+hk*8..+7] (matches A fragment).
__device__ __forceinline__ v16bf load_b_frag(const __bf16* rowN, int hk) {
    v8bf lo = *(const v8bf*)(rowN + hk * 8);
    v8bf hi = *(const v8bf*)(rowN + 16 + hk * 8);
    return cat8(lo, hi);
}

__device__ __forceinline__ v8f wmma_bf16(v16bf a, v16bf b, v8f c) {
    return __builtin_amdgcn_wmma_f32_16x16x32_bf16(false, a, false, b,
                                                   (short)0, c, false, false);
}

// ---------- conversion / packing ----------

__global__ __launch_bounds__(256)
void f32_to_bf16_kernel(const float* __restrict__ src, __bf16* __restrict__ dst, int n) {
    int i = blockIdx.x * 256 + threadIdx.x;
    if (i < n) dst[i] = f2bf(src[i]);
}

// Build Xcat[8192][1536] = [sp | om1 | om2] in bf16.
__global__ __launch_bounds__(256)
void build_xcat_kernel(const float* __restrict__ sp, const float* __restrict__ om1,
                       const float* __restrict__ om2, __bf16* __restrict__ xcat) {
    int i = blockIdx.x * 256 + threadIdx.x;
    if (i >= N_ROWS * IN_CH) return;
    int row = i >> 9, c = i & (IN_CH - 1);
    size_t b = (size_t)row * (3 * IN_CH);
    xcat[b + c]             = f2bf(sp[i]);
    xcat[b + IN_CH + c]     = f2bf(om1[i]);
    xcat[b + 2 * IN_CH + c] = f2bf(om2[i]);
}

// ---------- GEMM: C[M,Nout] = X[:,xoff:xoff+K] @ W[:,woff:woff+K]^T + bias ----------
// One wave computes a 32x64 output tile: 2 A-frags x 4 B-frags = 8 WMMAs / 32-K-step.
// K loop unrolled by 2 with ping-pong fragment buffers: phase p consumes buffer p
// while loading buffer p^1 — zero register copies, VMEM overlaps the matrix pipe.
// Requires K % 64 == 0 (all call sites: 512/1024/1536).
// mode: 0 = store f32 (optionally accumulate), 1 = bf16 row-major, 2 = bf16 transposed.
__global__ __launch_bounds__(256)
void gemm_xw_bf16(const __bf16* __restrict__ X, int ldx, int xoff,
                  const __bf16* __restrict__ W, int ldw, int woff,
                  const float* __restrict__ bias,
                  int K, int M, int Nout,
                  float* __restrict__ outF,
                  __bf16* __restrict__ outB,
                  __bf16* __restrict__ outBT, int ldT,
                  int mode, int accumulate)
{
    const int lane = threadIdx.x & 31;
    const int wid  = threadIdx.x >> 5;
    const int ntiles64 = Nout >> 6;
    const int gw = blockIdx.x * 8 + wid;
    if (gw >= (M >> 5) * ntiles64) return;
    const int mbase = (gw / ntiles64) << 5;
    const int nbase = (gw % ntiles64) << 6;
    const int m  = lane & 15;       // also the B-fragment column index n
    const int hk = lane >> 4;

    const __bf16* arow0 = X + (size_t)(mbase + m) * ldx + xoff;
    const __bf16* arow1 = X + (size_t)(mbase + 16 + m) * ldx + xoff;
    const __bf16* brow[4];
#pragma unroll
    for (int j = 0; j < 4; ++j)
        brow[j] = W + (size_t)(nbase + j * 16 + m) * ldw + woff;

    v8f acc[2][4] = {};
    v16bf aF[2][2], bF[2][4];
    aF[0][0] = load_a_frag(arow0, hk);
    aF[0][1] = load_a_frag(arow1, hk);
#pragma unroll
    for (int j = 0; j < 4; ++j) bF[0][j] = load_b_frag(brow[j], hk);

    for (int kk = 0; kk < K; kk += 64) {
        // phase 0: load buffer 1 @ kk+32 (always in range: K % 64 == 0), use buffer 0
        {
            const int kn = kk + 32;
            aF[1][0] = load_a_frag(arow0 + kn, hk);
            aF[1][1] = load_a_frag(arow1 + kn, hk);
#pragma unroll
            for (int j = 0; j < 4; ++j) bF[1][j] = load_b_frag(brow[j] + kn, hk);
#pragma unroll
            for (int i = 0; i < 2; ++i)
#pragma unroll
                for (int j = 0; j < 4; ++j)
                    acc[i][j] = wmma_bf16(aF[0][i], bF[0][j], acc[i][j]);
        }
        // phase 1: load buffer 0 @ kk+64 (if any), use buffer 1
        {
            const int kn = kk + 64;
            if (kn < K) {
                aF[0][0] = load_a_frag(arow0 + kn, hk);
                aF[0][1] = load_a_frag(arow1 + kn, hk);
#pragma unroll
                for (int j = 0; j < 4; ++j) bF[0][j] = load_b_frag(brow[j] + kn, hk);
            }
#pragma unroll
            for (int i = 0; i < 2; ++i)
#pragma unroll
                for (int j = 0; j < 4; ++j)
                    acc[i][j] = wmma_bf16(aF[1][i], bF[1][j], acc[i][j]);
        }
    }

#pragma unroll
    for (int i = 0; i < 2; ++i) {
#pragma unroll
        for (int j = 0; j < 4; ++j) {
            int chan = nbase + j * 16 + m;
            float badd = bias ? bias[chan] : 0.0f;
#pragma unroll
            for (int r = 0; r < 8; ++r) {
                int row = mbase + i * 16 + r + 8 * hk;   // D: M = r + 8*(lane>>4)
                float v = acc[i][j][r] + badd;
                if (mode == 0) {
                    size_t idx = (size_t)row * Nout + chan;
                    if (accumulate) v += outF[idx];
                    outF[idx] = v;
                } else if (mode == 1) {
                    outB[(size_t)row * Nout + chan] = f2bf(v);
                } else {
                    outBT[(size_t)chan * ldT + row] = f2bf(v);
                }
            }
        }
    }
}

// ---------- flash attention ----------
// Block: 256 threads = 8 waves, 16 query rows. Per 128-key iteration:
// wave w owns the score tile for keys [w*16, w*16+16) and output channels [w*64, w*64+64).
#define QSTRIDE 520   // 512 + 8 bf16 pad (keeps 16B alignment, breaks bank conflicts)
#define PSTRIDE 136   // 128 + 8 bf16 pad

__global__ __launch_bounds__(256)
void flash_attn_kernel(const __bf16* __restrict__ Q,
                       const __bf16* __restrict__ Km,
                       const __bf16* __restrict__ Vt,   // [512][8192] bf16
                       float* __restrict__ Out,
                       float scale)
{
    __shared__ __bf16 qlds[16 * QSTRIDE];
    __shared__ __bf16 plds[16 * PSTRIDE];
    __shared__ float red_max[8][16];
    __shared__ float red_sum[8][16];
    __shared__ float mnewS[16], factS[16], m_run[16], l_run[16];

    const int tid  = threadIdx.x;
    const int lane = tid & 31;
    const int wid  = tid >> 5;
    const int m    = lane & 15;
    const int hk   = lane >> 4;
    const int mbase = blockIdx.x << 4;
    const int chanBase = wid * 64;

    // Stage this block's 16 q rows into LDS (bf16).
    for (int i = tid; i < 16 * IN_CH; i += 256) {
        int row = i >> 9, c = i & (IN_CH - 1);
        qlds[row * QSTRIDE + c] = Q[(size_t)(mbase + row) * IN_CH + c];
    }
    if (tid < 16) { m_run[tid] = -__builtin_inff(); l_run[tid] = 0.0f; }
    __syncthreads();

    v8f acc[4] = {};

    for (int t = 0; t < N_ROWS / 128; ++t) {
        const int kb = t * 128 + wid * 16;          // this wave's 16 keys

        // ---- S tile = Q Kt (16x16): K loop unrolled x2, ping-pong buffers ----
        v8f s = {};
        const __bf16* krow = Km + (size_t)(kb + m) * IN_CH;
        const __bf16* qrow = qlds + m * QSTRIDE;
        if (t + 1 < N_ROWS / 128) {                 // prefetch next tile's K/V rows
            __builtin_prefetch(Km + (size_t)(kb + 128 + m) * IN_CH, 0, 1);
            __builtin_prefetch(Vt + (size_t)(chanBase + m) * N_ROWS + (t + 1) * 128, 0, 1);
        }
        v16bf qa[2], kf[2];
        qa[0] = load_a_frag(qrow, hk);
        kf[0] = load_b_frag(krow, hk);
        for (int kk = 0; kk < IN_CH; kk += 64) {
            qa[1] = load_a_frag(qrow + kk + 32, hk);
            kf[1] = load_b_frag(krow + kk + 32, hk);
            s = wmma_bf16(qa[0], kf[0], s);
            if (kk + 64 < IN_CH) {
                qa[0] = load_a_frag(qrow + kk + 64, hk);
                kf[0] = load_b_frag(krow + kk + 64, hk);
            }
            s = wmma_bf16(qa[1], kf[1], s);
        }

        // ---- row max within tile (reduce over 16 lanes per half) ----
        float mx[8];
#pragma unroll
        for (int r = 0; r < 8; ++r) {
            float v = s[r] * scale;
            s[r] = v;
#pragma unroll
            for (int sh = 8; sh >= 1; sh >>= 1)
                v = fmaxf(v, __shfl_xor(v, sh, 32));
            mx[r] = v;
        }
        if (m == 0) {
#pragma unroll
            for (int r = 0; r < 8; ++r) red_max[wid][r + 8 * hk] = mx[r];
        }
        __syncthreads();                                       // b1

        if (tid < 16) {
            float mo = m_run[tid];
            float v = mo;
#pragma unroll
            for (int w = 0; w < 8; ++w) v = fmaxf(v, red_max[w][tid]);
            mnewS[tid] = v;
            factS[tid] = __expf(mo - v);                       // 0 on first iter
            m_run[tid] = v;
        }
        __syncthreads();                                       // b2

        // ---- rescale acc, exponentiate, emit P (bf16, LDS) + partial sums ----
        float fr[8];
#pragma unroll
        for (int r = 0; r < 8; ++r) fr[r] = factS[r + 8 * hk];
#pragma unroll
        for (int j = 0; j < 4; ++j)
#pragma unroll
            for (int r = 0; r < 8; ++r) acc[j][r] = acc[j][r] * fr[r];
#pragma unroll
        for (int r = 0; r < 8; ++r) {
            int row = r + 8 * hk;
            float p = __expf(s[r] - mnewS[row]);
            plds[row * PSTRIDE + wid * 16 + m] = f2bf(p);
            float sm = p;
#pragma unroll
            for (int sh = 8; sh >= 1; sh >>= 1) sm += __shfl_xor(sm, sh, 32);
            if (m == 0) red_sum[wid][row] = sm;
        }
        __syncthreads();                                       // b3

        if (tid < 16) {
            float sm = 0.0f;
#pragma unroll
            for (int w = 0; w < 8; ++w) sm += red_sum[w][tid];
            l_run[tid] = l_run[tid] * factS[tid] + sm;
        }

        // ---- O += P V : fully unrolled 4-step ping-pong (ks & 1 selects buffer) ----
        const int keyBase = t * 128;
        const __bf16* prow = plds + m * PSTRIDE;
        const __bf16* vrow[4];
#pragma unroll
        for (int j = 0; j < 4; ++j)
            vrow[j] = Vt + (size_t)(chanBase + j * 16 + m) * N_ROWS + keyBase;

        v16bf paF[2], vbF[2][4];
        paF[0] = load_a_frag(prow, hk);
#pragma unroll
        for (int j = 0; j < 4; ++j) vbF[0][j] = load_b_frag(vrow[j], hk);
#pragma unroll
        for (int ks = 0; ks < 4; ++ks) {
            const int cur = ks & 1, nxt = cur ^ 1;
            if (ks < 3) {
                paF[nxt] = load_a_frag(prow + (ks + 1) * 32, hk);
#pragma unroll
                for (int j = 0; j < 4; ++j)
                    vbF[nxt][j] = load_b_frag(vrow[j] + (ks + 1) * 32, hk);
            }
#pragma unroll
            for (int j = 0; j < 4; ++j)
                acc[j] = wmma_bf16(paF[cur], vbF[cur][j], acc[j]);
        }
    }
    __syncthreads();

#pragma unroll
    for (int j = 0; j < 4; ++j) {
        int chan = chanBase + j * 16 + m;
#pragma unroll
        for (int r = 0; r < 8; ++r) {
            int row = r + 8 * hk;
            Out[(size_t)(mbase + row) * OUT_CH + chan] = acc[j][r] / l_run[row];
        }
    }
}

// ---------- final blend: softmax over {sigmoid(g1), sigmoid(g2)} ----------
__global__ __launch_bounds__(256)
void combine_kernel(const float* __restrict__ g1, const float* __restrict__ g2,
                    const float* __restrict__ a1, const float* __restrict__ a2,
                    float* __restrict__ out, int n) {
    int i = blockIdx.x * 256 + threadIdx.x;
    if (i >= n) return;
    float c1 = 1.0f / (1.0f + __expf(-g1[i]));
    float c2 = 1.0f / (1.0f + __expf(-g2[i]));
    float w1 = 1.0f / (1.0f + __expf(c2 - c1));   // exp(c1)/(exp(c1)+exp(c2))
    out[i] = w1 * a1[i] + (1.0f - w1) * a2[i];
}

// ---------- host launch ----------

extern "C" void kernel_launch(void* const* d_in, const int* in_sizes, int n_in,
                              void* d_out, int out_size, void* d_ws, size_t ws_size,
                              hipStream_t stream) {
    const float* sp   = (const float*)d_in[0];
    const float* om1  = (const float*)d_in[1];
    const float* om2  = (const float*)d_in[2];
    const float* q_w  = (const float*)d_in[3];
    const float* q_b  = (const float*)d_in[4];
    const float* k1_w = (const float*)d_in[5];
    const float* k1_b = (const float*)d_in[6];
    const float* k2_w = (const float*)d_in[7];
    const float* k2_b = (const float*)d_in[8];
    const float* v_w  = (const float*)d_in[9];
    const float* v_b  = (const float*)d_in[10];
    const float* c1_w = (const float*)d_in[11];
    const float* c1_b = (const float*)d_in[12];

    char* ws = (char*)d_ws;
    size_t off = 0;
    auto alloc = [&](size_t bytes) -> void* {
        void* p = ws + off;
        off += (bytes + 255) & ~(size_t)255;
        return p;
    };

    __bf16* Xcat = (__bf16*)alloc((size_t)N_ROWS * 3 * IN_CH * 2);
    __bf16* Qb   = (__bf16*)alloc((size_t)N_ROWS * OUT_CH * 2);
    __bf16* K1b  = (__bf16*)alloc((size_t)N_ROWS * OUT_CH * 2);
    __bf16* K2b  = (__bf16*)alloc((size_t)N_ROWS * OUT_CH * 2);
    __bf16* Vtb  = (__bf16*)alloc((size_t)OUT_CH * N_ROWS * 2);
    __bf16* qwb  = (__bf16*)alloc((size_t)512 * 512 * 2);
    __bf16* k1wb = (__bf16*)alloc((size_t)512 * 512 * 2);
    __bf16* k2wb = (__bf16*)alloc((size_t)512 * 512 * 2);
    __bf16* vwb  = (__bf16*)alloc((size_t)512 * 1536 * 2);
    __bf16* c1wb = (__bf16*)alloc((size_t)512 * 1024 * 2);
    float*  g1   = (float*)alloc((size_t)N_ROWS * OUT_CH * 4);
    float*  g2   = (float*)alloc((size_t)N_ROWS * OUT_CH * 4);
    float*  at1  = (float*)alloc((size_t)N_ROWS * OUT_CH * 4);
    float*  at2  = (float*)alloc((size_t)N_ROWS * OUT_CH * 4);

    // 1) convert weights to bf16
    f32_to_bf16_kernel<<<(512*512 + 255)/256, 256, 0, stream>>>(q_w,  qwb,  512*512);
    f32_to_bf16_kernel<<<(512*512 + 255)/256, 256, 0, stream>>>(k1_w, k1wb, 512*512);
    f32_to_bf16_kernel<<<(512*512 + 255)/256, 256, 0, stream>>>(k2_w, k2wb, 512*512);
    f32_to_bf16_kernel<<<(512*1536 + 255)/256, 256, 0, stream>>>(v_w,  vwb,  512*1536);
    f32_to_bf16_kernel<<<(512*1024 + 255)/256, 256, 0, stream>>>(c1_w, c1wb, 512*1024);

    // 2) pack inputs into bf16 concat [sp|om1|om2]
    build_xcat_kernel<<<(N_ROWS*IN_CH + 255)/256, 256, 0, stream>>>(sp, om1, om2, Xcat);

    // 3) projections: 256 blocks = (8192/32 M-tiles) * (512/64 N-tiles) / 8 waves
    gemm_xw_bf16<<<256, 256, 0, stream>>>(Xcat, 1536, 0,    qwb,  512,  0,   q_b,
                                          512,  N_ROWS, OUT_CH, nullptr, Qb,  nullptr, 0, 1, 0);
    gemm_xw_bf16<<<256, 256, 0, stream>>>(Xcat, 1536, 512,  k1wb, 512,  0,   k1_b,
                                          512,  N_ROWS, OUT_CH, nullptr, K1b, nullptr, 0, 1, 0);
    gemm_xw_bf16<<<256, 256, 0, stream>>>(Xcat, 1536, 1024, k2wb, 512,  0,   k2_b,
                                          512,  N_ROWS, OUT_CH, nullptr, K2b, nullptr, 0, 1, 0);
    // v = [sp|om1|om2] @ v_w^T, stored transposed for the P@V B-fragments
    gemm_xw_bf16<<<256, 256, 0, stream>>>(Xcat, 1536, 0,    vwb,  1536, 0,   v_b,
                                          1536, N_ROWS, OUT_CH, nullptr, nullptr, Vtb, N_ROWS, 2, 0);
    // gate logits: g1 = [sp|om1]@c1^T ; g2 = sp@c1[:, :512]^T + om2@c1[:, 512:]^T
    gemm_xw_bf16<<<256, 256, 0, stream>>>(Xcat, 1536, 0,    c1wb, 1024, 0,   c1_b,
                                          1024, N_ROWS, OUT_CH, g1, nullptr, nullptr, 0, 0, 0);
    gemm_xw_bf16<<<256, 256, 0, stream>>>(Xcat, 1536, 0,    c1wb, 1024, 0,   c1_b,
                                          512,  N_ROWS, OUT_CH, g2, nullptr, nullptr, 0, 0, 0);
    gemm_xw_bf16<<<256, 256, 0, stream>>>(Xcat, 1536, 1024, c1wb, 1024, 512, nullptr,
                                          512,  N_ROWS, OUT_CH, g2, nullptr, nullptr, 0, 0, 1);

    // 4) two flash attentions
    const float scale = 0.044194173824159216f;   // 1/sqrt(512)
    flash_attn_kernel<<<N_ROWS/16, 256, 0, stream>>>(Qb, K1b, Vtb, at1, scale);
    flash_attn_kernel<<<N_ROWS/16, 256, 0, stream>>>(Qb, K2b, Vtb, at2, scale);

    // 5) blend
    combine_kernel<<<(N_ROWS*OUT_CH + 255)/256, 256, 0, stream>>>(
        g1, g2, at1, at2, (float*)d_out, N_ROWS * OUT_CH);
}
